// MultiHeadSelfAttention_76527727280327
// MI455X (gfx1250) — matrix operand order repaired and checked by reading, not compiled, not run on hardware
//
#include <hip/hip_runtime.h>
#include <hip/hip_bf16.h>
#include <math.h>

// Problem constants (B,T,C,H,D) = (4,2048,1024,16,64)
#define Bn 4
#define Tn 2048
#define Cn 1024
#define Hn 16
#define Dn 64

typedef __bf16 bf16t;
typedef bf16t v16bf __attribute__((ext_vector_type(16)));
typedef float v8f __attribute__((ext_vector_type(8)));

union FragBF { uint4 u[2]; unsigned int w[8]; v16bf v; };

__device__ __forceinline__ unsigned short f2bf(float f) {
    unsigned u = __float_as_uint(f);
    u += 0x7FFFu + ((u >> 16) & 1u);          // round-to-nearest-even
    return (unsigned short)(u >> 16);
}

__device__ __forceinline__ v8f wmma_bf16(v16bf a, v16bf b, v8f c) {
    // D = A(16x32 bf16) * B(32x16 bf16) + C(16x16 f32)
    return __builtin_amdgcn_wmma_f32_16x16x32_bf16(false, a, false, b, (short)0, c,
                                                   false, false);
}

// ---------------------------------------------------------------- converts
__global__ void k_conv_bf16(const float* __restrict__ in,
                            unsigned short* __restrict__ out, int n) {
    int i = blockIdx.x * blockDim.x + threadIdx.x;
    if (i < n) out[i] = f2bf(in[i]);
}

// W (Kd x Nd, row-major f32) -> Wt (Nd x Kd, row-major bf16)
__global__ void k_conv_transpose(const float* __restrict__ W,
                                 unsigned short* __restrict__ Wt,
                                 int Kd, int Nd) {
    int i = blockIdx.x * blockDim.x + threadIdx.x;
    if (i >= Kd * Nd) return;
    int n = i / Kd;
    int k = i - n * Kd;
    Wt[i] = f2bf(W[(size_t)k * Nd + n]);
}

__global__ void k_zero_f32(float* __restrict__ p, int n) {
    int i = blockIdx.x * blockDim.x + threadIdx.x;
    if (i < n) p[i] = 0.f;
}

// ------------------------------------------------------------ GEMM core
// One wave computes a 16x64 f32 tile over K=1024, software-pipelined:
// iteration k+1's fragments are in flight while iteration k's WMMAs execute.
// A: M x 1024 bf16 row-major.  Bt: N x 1024 bf16 row-major (pre-transposed B).
__device__ __forceinline__ void wmma_rowtile_1024(const unsigned short* __restrict__ Arow,
                                                  const unsigned short* __restrict__ BtBase,
                                                  int lane, v8f acc[4]) {
    const int hw = lane >> 4;       // half-wave (0/1)
    const int ln = lane & 15;
    const unsigned short* ap = Arow + hw * 16;
    const unsigned short* bp = BtBase + (size_t)ln * 1024 + hw * 16;

    // A layout: elems j -> k = kc + hw*16 + j (contiguous)
    // B layout: lane col n = ln, elems j -> k = kc + hw*16 + j (contiguous)
    FragBF a, bb[4];
    a.u[0] = *(const uint4*)(ap);
    a.u[1] = *(const uint4*)(ap + 8);
    #pragma unroll
    for (int s = 0; s < 4; ++s) {
        bb[s].u[0] = *(const uint4*)(bp + (size_t)s * 16 * 1024);
        bb[s].u[1] = *(const uint4*)(bp + (size_t)s * 16 * 1024 + 8);
    }

    #pragma unroll 4
    for (int kc = 0; kc < 1024; kc += 32) {
        FragBF an, bn[4];
        const int kn = kc + 32;
        if (kn < 1024) {                       // prefetch next iteration
            an.u[0] = *(const uint4*)(ap + kn);
            an.u[1] = *(const uint4*)(ap + kn + 8);
            #pragma unroll
            for (int s = 0; s < 4; ++s) {
                bn[s].u[0] = *(const uint4*)(bp + (size_t)s * 16 * 1024 + kn);
                bn[s].u[1] = *(const uint4*)(bp + (size_t)s * 16 * 1024 + kn + 8);
            }
        }
        #pragma unroll
        for (int s = 0; s < 4; ++s)
            acc[s] = wmma_bf16(a.v, bb[s].v, acc[s]);
        a = an;
        #pragma unroll
        for (int s = 0; s < 4; ++s) bb[s] = bn[s];
    }
}

// qkv = xh @ wqkvT^T ; scatter into Q(B,H,T,D), K(B,H,T,D), Vt(B,H,D,T) bf16.
// Q is pre-scaled by 1/sqrt(D).
__global__ void __launch_bounds__(256)
k_qkv_gemm(const unsigned short* __restrict__ xh,
           const unsigned short* __restrict__ wT,
           unsigned short* __restrict__ Qb,
           unsigned short* __restrict__ Kb,
           unsigned short* __restrict__ Vt) {
    const int wid  = (blockIdx.x * blockDim.x + threadIdx.x) >> 5;
    const int lane = threadIdx.x & 31;
    const int hw = lane >> 4, ln = lane & 15;
    const int r = wid / 48, cg = wid % 48;     // 512 row tiles x 48 col groups

    v8f z = {0.f, 0.f, 0.f, 0.f, 0.f, 0.f, 0.f, 0.f};
    v8f acc[4] = {z, z, z, z};
    wmma_rowtile_1024(xh + (size_t)(r * 16 + ln) * 1024,
                      wT + (size_t)cg * 64 * 1024, lane, acc);

    const int sel = cg >> 4;                   // 0=q 1=k 2=v (64-wide group == one head)
    const int h   = cg & 15;
    const int rowb = r * 16;
    const int b  = rowb >> 11;                 // / T
    const int t0 = (rowb & (Tn - 1)) + hw * 8;
    const float qs = (sel == 0) ? 0.125f : 1.0f;   // 1/sqrt(64) folded into Q

    if (sel < 2) {
        unsigned short* dst = (sel == 0 ? Qb : Kb) + (size_t)(b * Hn + h) * Tn * Dn;
        #pragma unroll
        for (int s = 0; s < 4; ++s) {
            const int d = s * 16 + ln;
            #pragma unroll
            for (int j = 0; j < 8; ++j)
                dst[(size_t)(t0 + j) * Dn + d] = f2bf(acc[s][j] * qs);
        }
    } else {
        #pragma unroll
        for (int s = 0; s < 4; ++s) {
            const int d = s * 16 + ln;
            union { unsigned short s16[8]; uint4 u; } pk;
            #pragma unroll
            for (int j = 0; j < 8; ++j) pk.s16[j] = f2bf(acc[s][j]);
            *(uint4*)(Vt + ((size_t)(b * Hn + h) * Dn + d) * Tn + t0) = pk.u;
        }
    }
}

// out = Oh @ wprojT^T, f32 result straight to d_out.
__global__ void __launch_bounds__(256)
k_proj_gemm(const unsigned short* __restrict__ Oh,
            const unsigned short* __restrict__ wT,
            float* __restrict__ outp) {
    const int wid  = (blockIdx.x * blockDim.x + threadIdx.x) >> 5;
    const int lane = threadIdx.x & 31;
    const int hw = lane >> 4, ln = lane & 15;
    const int r = wid / 16, cg = wid % 16;     // 512 row tiles x 16 col groups

    v8f z = {0.f, 0.f, 0.f, 0.f, 0.f, 0.f, 0.f, 0.f};
    v8f acc[4] = {z, z, z, z};
    wmma_rowtile_1024(Oh + (size_t)(r * 16 + ln) * 1024,
                      wT + (size_t)cg * 64 * 1024, lane, acc);

    const int row0 = r * 16 + hw * 8;
    #pragma unroll
    for (int s = 0; s < 4; ++s) {
        const int c = cg * 64 + s * 16 + ln;
        #pragma unroll
        for (int j = 0; j < 8; ++j)
            outp[(size_t)(row0 + j) * Cn + c] = acc[s][j];
    }
}

// K-tile B-fragments (two K=32 d-chunks) for the Q@K^T score WMMA.
__device__ __forceinline__ void load_ktile(const unsigned short* __restrict__ Kbh,
                                           int kb, int hw, int ln,
                                           FragBF& b0, FragBF& b1) {
    const unsigned short* krow = Kbh + (size_t)(kb * 16 + ln) * Dn;
    b0.u[0] = *(const uint4*)(krow + hw * 16);
    b0.u[1] = *(const uint4*)(krow + hw * 16 + 8);
    b1.u[0] = *(const uint4*)(krow + 32 + hw * 16);
    b1.u[1] = *(const uint4*)(krow + 32 + hw * 16 + 8);
}

// ------------------------------------------------------------ attention
// One block = one (b, 16-query tile). 16 waves = 16 heads in lockstep.
// Two-pass softmax; attn_avg reduced across heads in LDS (ds_add_f32),
// written to HBM exactly once. K/V fragment loads are prefetched so global
// latency overlaps the softmax VALU work.
__global__ void __launch_bounds__(512)
k_attn(const unsigned short* __restrict__ Qb,
       const unsigned short* __restrict__ Kb,
       const unsigned short* __restrict__ Vt,
       const unsigned char* __restrict__ kpm,
       unsigned short* __restrict__ Oh,
       float* __restrict__ avg) {
    __shared__ unsigned int Pp[Hn][256];   // per-wave P tile, A-layout pairs (16 kp x 16 m)
    __shared__ float avgbuf[16 * 32];      // 16 queries x 32 keys, head-reduced

    const int qt = blockIdx.x & 127;
    const int b  = blockIdx.x >> 7;
    const int h  = threadIdx.x >> 5;       // wave id == head
    const int lane = threadIdx.x & 31;
    const int hw = lane >> 4, ln = lane & 15;
    const int qbase = qt * 16;
    const float slope = exp2f(-0.5f * (float)(h + 1));   // ALiBi slope for H=16
    const unsigned char* kpmb = kpm + (size_t)b * Tn;

    avgbuf[threadIdx.x] = 0.f;             // 512 threads, 512 cells
    __syncthreads();

    const unsigned short* Qbh = Qb + (size_t)(b * Hn + h) * Tn * Dn;
    const unsigned short* Kbh = Kb + (size_t)(b * Hn + h) * Tn * Dn;
    const unsigned short* Vbh = Vt + (size_t)(b * Hn + h) * Dn * Tn;

    // Q A-fragments over D=64 (two K=32 chunks); Q already scaled by 1/8.
    FragBF qa0, qa1;
    {
        const unsigned short* qrow = Qbh + (size_t)(qbase + ln) * Dn;
        qa0.u[0] = *(const uint4*)(qrow + hw * 16);
        qa0.u[1] = *(const uint4*)(qrow + hw * 16 + 8);
        qa1.u[0] = *(const uint4*)(qrow + 32 + hw * 16);
        qa1.u[1] = *(const uint4*)(qrow + 32 + hw * 16 + 8);
    }

    const int nkt = qt + 1;                // causal: key tiles 0..qt
    float Mr[8], Sr[8];
    #pragma unroll
    for (int j = 0; j < 8; ++j) { Mr[j] = -1e30f; Sr[j] = 0.f; }

    // ---------- pass 1: row max / sum (K tiles prefetched one ahead)
    {
        FragBF ck0, ck1;
        load_ktile(Kbh, 0, hw, ln, ck0, ck1);
        for (int kb = 0; kb < nkt; ++kb) {
            v8f sa = {0.f, 0.f, 0.f, 0.f, 0.f, 0.f, 0.f, 0.f};
            sa = wmma_bf16(qa0.v, ck0.v, sa);
            sa = wmma_bf16(qa1.v, ck1.v, sa);
            FragBF nk0, nk1;
            if (kb + 1 < nkt) load_ktile(Kbh, kb + 1, hw, ln, nk0, nk1);

            const int kj = kb * 16 + ln;
            const int pad = (int)kpmb[kj];
            #pragma unroll
            for (int j = 0; j < 8; ++j) {
                const int qi = qbase + hw * 8 + j;
                float s = sa[j] - slope * fmaxf((float)(qi - kj), 0.f);
                if (kj > qi || pad) s = -1e30f;
                float mv = s;                                  // 16-lane (half-wave) max
                mv = fmaxf(mv, __shfl_xor(mv, 1, 32));
                mv = fmaxf(mv, __shfl_xor(mv, 2, 32));
                mv = fmaxf(mv, __shfl_xor(mv, 4, 32));
                mv = fmaxf(mv, __shfl_xor(mv, 8, 32));
                const float Mn = fmaxf(Mr[j], mv);
                float e = __expf(s - Mn);                      // 16-lane sum
                e += __shfl_xor(e, 1, 32);
                e += __shfl_xor(e, 2, 32);
                e += __shfl_xor(e, 4, 32);
                e += __shfl_xor(e, 8, 32);
                Sr[j] = Sr[j] * __expf(Mr[j] - Mn) + e;
                Mr[j] = Mn;
            }
            ck0 = nk0;
            ck1 = nk1;
        }
    }
    float rS[8];
    #pragma unroll
    for (int j = 0; j < 8; ++j) rS[j] = 1.f / Sr[j];

    // ---------- pass 2: probs -> attn_avg + P@V
    v8f z = {0.f, 0.f, 0.f, 0.f, 0.f, 0.f, 0.f, 0.f};
    v8f oacc[4] = {z, z, z, z};
    const int npair = (nkt + 1) >> 1;
    unsigned short* ppw = (unsigned short*)Pp[h];

    FragBF ck0, ck1;
    load_ktile(Kbh, 0, hw, ln, ck0, ck1);    // prefetch first K tile

    for (int pr = 0; pr < npair; ++pr) {
        const int kb2 = pr * 32;

        // issue V-fragment loads for this pair early; consumed after softmax
        FragBF vb[4];
        #pragma unroll
        for (int dc = 0; dc < 4; ++dc) {
            const unsigned short* vrow = Vbh + (size_t)(dc * 16 + ln) * Tn + kb2 + hw * 16;
            vb[dc].u[0] = *(const uint4*)(vrow);
            vb[dc].u[1] = *(const uint4*)(vrow + 8);
        }

        #pragma unroll
        for (int sub = 0; sub < 2; ++sub) {
            const int kb  = pr * 2 + sub;
            const int kip = sub * 16 + ln;        // key index inside the 32-key pair
            const int kp  = kip >> 1, par = kip & 1;
            if (kb < nkt) {
                v8f sa = {0.f, 0.f, 0.f, 0.f, 0.f, 0.f, 0.f, 0.f};
                sa = wmma_bf16(qa0.v, ck0.v, sa);
                sa = wmma_bf16(qa1.v, ck1.v, sa);
                FragBF nk0, nk1;
                if (kb + 1 < nkt) load_ktile(Kbh, kb + 1, hw, ln, nk0, nk1);

                const int kj = kb * 16 + ln;
                const int pad = (int)kpmb[kj];
                #pragma unroll
                for (int j = 0; j < 8; ++j) {
                    const int qi = qbase + hw * 8 + j;
                    float s = sa[j] - slope * fmaxf((float)(qi - kj), 0.f);
                    float p = (kj > qi || pad) ? 0.f : __expf(s - Mr[j]) * rS[j];
                    ppw[(kp * 16 + hw * 8 + j) * 2 + par] = f2bf(p);
                    atomicAdd(&avgbuf[(hw * 8 + j) * 32 + kip], p);   // ds_add_f32
                }
                ck0 = nk0;
                ck1 = nk1;
            } else {
                #pragma unroll
                for (int j = 0; j < 8; ++j)
                    ppw[(kp * 16 + hw * 8 + j) * 2 + par] = 0;
            }
        }

        // A-fragment of P from LDS (same-wave LDS ops are in-order)
        FragBF pa;
        #pragma unroll
        for (int dw = 0; dw < 8; ++dw) {
            const int kp2 = dw + hw * 8;
            pa.w[dw] = Pp[h][kp2 * 16 + ln];
        }
        // O += P(16x32) @ V(32x16) for each 16-wide d chunk
        #pragma unroll
        for (int dc = 0; dc < 4; ++dc)
            oacc[dc] = wmma_bf16(pa.v, vb[dc].v, oacc[dc]);

        // flush head-reduced 16x32 attn_avg tile, once per pair
        __syncthreads();
        {
            const int qr = threadIdx.x >> 5;   // 0..15
            const int kc = threadIdx.x & 31;   // 0..31
            const float v = avgbuf[qr * 32 + kc];
            avg[((size_t)b * Tn + qbase + qr) * Tn + kb2 + kc] = v * (1.f / (float)Hn);
            avgbuf[qr * 32 + kc] = 0.f;
        }
        __syncthreads();
    }

    // store O (already normalized) as bf16 into (B,T,C) for the proj GEMM
    #pragma unroll
    for (int dc = 0; dc < 4; ++dc) {
        const int col = h * Dn + dc * 16 + ln;
        #pragma unroll
        for (int j = 0; j < 8; ++j) {
            const int t = qbase + hw * 8 + j;
            Oh[((size_t)b * Tn + t) * Cn + col] = f2bf(oacc[dc][j]);
        }
    }
}

// ---------------------------------------------------------------- launch
extern "C" void kernel_launch(void* const* d_in, const int* in_sizes, int n_in,
                              void* d_out, int out_size, void* d_ws, size_t ws_size,
                              hipStream_t stream) {
    const float* x        = (const float*)d_in[0];
    const float* w_qkv    = (const float*)d_in[1];
    const float* w_proj   = (const float*)d_in[2];
    const unsigned char* kpm = (const unsigned char*)d_in[3];   // bool (B,T)

    float* outp = (float*)d_out;                         // (B,T,C) f32
    float* avg  = outp + (size_t)Bn * Tn * Cn;           // (B,T,T) f32

    // workspace layout (bf16 halves), ~92 MB total
    unsigned short* xh     = (unsigned short*)d_ws;
    unsigned short* wqkvT  = xh     + (size_t)Bn * Tn * Cn;       // (3C, C)
    unsigned short* wprojT = wqkvT  + (size_t)3 * Cn * Cn;        // (C, C)
    unsigned short* Qb     = wprojT + (size_t)Cn * Cn;            // (B,H,T,D)
    unsigned short* Kb     = Qb     + (size_t)Bn * Hn * Tn * Dn;  // (B,H,T,D)
    unsigned short* Vt     = Kb     + (size_t)Bn * Hn * Tn * Dn;  // (B,H,D,T)
    unsigned short* Oh     = Vt     + (size_t)Bn * Hn * Tn * Dn;  // (B,T,C)

    const int nx = Bn * Tn * Cn;
    k_conv_bf16<<<(nx + 255) / 256, 256, 0, stream>>>(x, xh, nx);

    const int nq = 3 * Cn * Cn;
    k_conv_transpose<<<(nq + 255) / 256, 256, 0, stream>>>(w_qkv, wqkvT, Cn, 3 * Cn);

    const int npj = Cn * Cn;
    k_conv_transpose<<<(npj + 255) / 256, 256, 0, stream>>>(w_proj, wprojT, Cn, Cn);

    const int na = Bn * Tn * Tn;
    k_zero_f32<<<(na + 255) / 256, 256, 0, stream>>>(avg, na);

    // QKV GEMM: (B*T/16)=512 row tiles x (3C/64)=48 col groups, 1 wave each
    k_qkv_gemm<<<512 * 48 / 8, 256, 0, stream>>>(xh, wqkvT, Qb, Kb, Vt);

    // attention: one block per (b, 16-query tile), 16 waves = 16 heads
    k_attn<<<Bn * (Tn / 16), 512, 0, stream>>>(Qb, Kb, Vt, kpm, Oh, avg);

    // output projection: 512 row tiles x 16 col groups
    k_proj_gemm<<<512 * 16 / 8, 256, 0, stream>>>(Oh, wprojT, outp);
}
